// ImprovedStateEncoder_27453430956622
// MI455X (gfx1250) — compile-verified, matrix operand order checked by reading, not run on hardware
//
#include <hip/hip_runtime.h>
#include <hip/hip_bf16.h>

typedef __attribute__((ext_vector_type(16))) __bf16 v16bf;
typedef __attribute__((ext_vector_type(8)))  float  v8f;

#define D      256
#define DNODE  128
#define NLAYERS 3
#define BN_EPS 1e-5f

// ---------------------------------------------------------------------------
// Pack B [K,256] f32 (row-major) -> bf16 in WMMA B-fragment lane order.
// Bp[((chunk*256 + n)*32) + i], i in [0,32):
//   half = i>>4 selects lane group (lane<16 vs lane>=16),
//   j = i&15 -> k_local = (j<8 ? j : j+8) + half*8  (per ISA 16-bit layout).
// Each lane loads its 16 bf16 values (32 B) with one aligned v16bf load.
// ---------------------------------------------------------------------------
__global__ void pack_b_kernel(const float* __restrict__ B, __bf16* __restrict__ Bp, int K) {
  int tid  = blockIdx.x * blockDim.x + threadIdx.x;   // K*256 threads
  int i    = tid & 31;
  int n    = (tid >> 5) & 255;
  int ch   = tid >> 13;                               // 256*32 = 8192 per chunk
  int half = i >> 4, j = i & 15;
  int kl   = (j < 8 ? j : j + 8) + half * 8;
  int k    = ch * 32 + kl;
  Bp[tid]  = (__bf16)B[(size_t)k * D + n];
}

// ---------------------------------------------------------------------------
// C[M,256] = A[M,K] @ B[K,256] + bias   (bf16 WMMA, f32 accumulate)
// Block = 256 threads = 8 waves; block owns 16 rows; wave w owns columns
// [16w,16w+16) and [128+16w,...): A staged once per block, reused over all
// 256 output columns (A read exactly once from HBM).
// K is a template param (128/256/512): fully unrolled, constant strides.
// Staging: 64-wide K chunks, one float4 (b128) per thread, double-buffered
// LDS (one barrier per 64 K), rows padded to 68 floats (bank-stagger by 4).
// MODE: 0 = store, 1 = relu+store, 2 = atomicAdd into C[scat[row]] (scatter)
// ---------------------------------------------------------------------------
template<int MODE, int K>
__global__ void gemm_wmma_n256(const float*  __restrict__ A,
                               const __bf16* __restrict__ Bp,
                               const float*  __restrict__ bias,
                               float*        __restrict__ C,
                               const int*    __restrict__ scat) {
  constexpr int STAGES = K / 64;
  __shared__ float As[2][16][68];                      // 68 = 64 + 4 pad (bank stagger)
  const int t     = threadIdx.x;
  const int lane  = t & 31;
  const int wave  = t >> 5;
  const int half  = lane >> 4;
  const int lmod  = lane & 15;
  const int mbase = blockIdx.x * 16;
  const int col0  = wave * 16 + lmod;
  const int col1  = col0 + 128;

  // staging: thread t loads float4 at row rs = t>>4, cols cs..cs+3
  const int rs = t >> 4;
  const int cs = (t & 15) * 4;
  const float* aRow = A + (size_t)(mbase + rs) * K + cs;

  v8f c0 = {0.f,0.f,0.f,0.f,0.f,0.f,0.f,0.f};
  v8f c1 = {0.f,0.f,0.f,0.f,0.f,0.f,0.f,0.f};

  *reinterpret_cast<float4*>(&As[0][rs][cs]) = *reinterpret_cast<const float4*>(aRow);
  __syncthreads();

#pragma unroll
  for (int s = 0; s < STAGES; ++s) {
    if (s + 1 < STAGES) {                              // prefetch next stage into other buffer
      *reinterpret_cast<float4*>(&As[(s + 1) & 1][rs][cs]) =
          *reinterpret_cast<const float4*>(aRow + (s + 1) * 64);
      if (s + 2 < STAGES)                              // CDNA5 global_prefetch_b8
        __builtin_prefetch(aRow + (s + 2) * 64, 0, 0);
    }
#pragma unroll
    for (int kk = 0; kk < 2; ++kk) {                   // two 32-wide WMMA chunks per stage
      const float4* rp =
          reinterpret_cast<const float4*>(&As[s & 1][lmod][kk * 32 + half * 8]);
      float4 f0 = rp[0], f1 = rp[1], f2 = rp[4], f3 = rp[5];
      float av[16] = {f0.x,f0.y,f0.z,f0.w, f1.x,f1.y,f1.z,f1.w,
                      f2.x,f2.y,f2.z,f2.w, f3.x,f3.y,f3.z,f3.w};
      v16bf a;
#pragma unroll
      for (int i = 0; i < 16; ++i) a[i] = (__bf16)av[i];

      const int ch = s * 2 + kk;
      const v16bf b0 = *reinterpret_cast<const v16bf*>(
          Bp + ((size_t)(ch * 256 + col0) * 32 + half * 16));
      const v16bf b1 = *reinterpret_cast<const v16bf*>(
          Bp + ((size_t)(ch * 256 + col1) * 32 + half * 16));

      c0 = __builtin_amdgcn_wmma_f32_16x16x32_bf16(false, a, false, b0, (short)0, c0, false, false);
      c1 = __builtin_amdgcn_wmma_f32_16x16x32_bf16(false, a, false, b1, (short)0, c1, false, false);
    }
    __syncthreads();
  }

  const float bv0 = bias[col0];
  const float bv1 = bias[col1];
#pragma unroll
  for (int r = 0; r < 8; ++r) {                        // C/D layout: VGPR r -> M = r (+8 hi lanes)
    int   mrow = mbase + r + half * 8;
    float v0   = c0[r] + bv0;
    float v1   = c1[r] + bv1;
    if (MODE == 1) { v0 = fmaxf(v0, 0.f); v1 = fmaxf(v1, 0.f); }
    if (MODE == 2) {
      int row = scat[mrow];
      atomicAdd(&C[(size_t)row * D + col0], v0);       // L2-resident destination
      atomicAdd(&C[(size_t)row * D + col1], v1);
    } else {
      C[(size_t)mrow * D + col0] = v0;
      C[(size_t)mrow * D + col1] = v1;
    }
  }
}

// xc[i] = [h[i], edge_agg[i]]; acc initialized to xc (GIN self term, eps=0)
__global__ void build_xc_kernel(const float* __restrict__ h, const float* __restrict__ ea,
                                float* __restrict__ xc, float* __restrict__ acc) {
  int i = blockIdx.x, c = threadIdx.x;
  float hv = h[(size_t)i * D + c];
  float ev = ea[(size_t)i * D + c];
  size_t b = (size_t)i * 2 * D;
  xc[b + c] = hv;  xc[b + D + c] = ev;
  acc[b + c] = hv; acc[b + D + c] = ev;
}

// acc[dst[e]] += xc[src[e]]  (512 floats per edge, float4 granularity)
__global__ void gin_msg_kernel(const float* __restrict__ xc, const int* __restrict__ src,
                               const int* __restrict__ dst, float* __restrict__ acc) {
  int tid = blockIdx.x * blockDim.x + threadIdx.x;     // E*128 threads
  int e = tid >> 7;
  int q = tid & 127;
  int s = src[e], d = dst[e];
  const float4 v = reinterpret_cast<const float4*>(xc)[(size_t)s * 128 + q];
  float* ap = acc + (size_t)d * 512 + q * 4;
  atomicAdd(ap + 0, v.x); atomicAdd(ap + 1, v.y);
  atomicAdd(ap + 2, v.z); atomicAdd(ap + 3, v.w);
}

// Per-column sum / sum-of-squares over z [N,256]
__global__ void colstat_kernel(const float* __restrict__ z,
                               float* __restrict__ s1, float* __restrict__ s2, int N) {
  int c = threadIdx.x;
  float s = 0.f, q = 0.f;
  for (int r = blockIdx.x; r < N; r += gridDim.x) {
    float v = z[(size_t)r * D + c];
    s += v; q += v * v;
  }
  atomicAdd(&s1[c], s);
  atomicAdd(&s2[c], q);
}

// h += relu(gamma * (z - mu) * rsqrt(var + eps) + beta), biased var (matches ref)
__global__ void bn_apply_kernel(const float* __restrict__ z,
                                const float* __restrict__ s1, const float* __restrict__ s2,
                                const float* __restrict__ gamma, const float* __restrict__ beta,
                                float* __restrict__ h, int N) {
  int i = blockIdx.x, c = threadIdx.x;
  float mu  = s1[c] / (float)N;
  float var = s2[c] / (float)N - mu * mu;
  float zn  = gamma[c] * (z[(size_t)i * D + c] - mu) * rsqrtf(var + BN_EPS) + beta[c];
  h[(size_t)i * D + c] += fmaxf(zn, 0.f);
}

__device__ __forceinline__ unsigned fenc(float f) {     // order-preserving float->uint
  unsigned u = __float_as_uint(f);
  return (u & 0x80000000u) ? ~u : (u | 0x80000000u);
}
__device__ __forceinline__ float fdec(unsigned u) {
  unsigned v = (u & 0x80000000u) ? (u & 0x7FFFFFFFu) : ~u;
  return __uint_as_float(v);
}

__global__ void pool_kernel(const float* __restrict__ h,
                            float* __restrict__ psum, unsigned* __restrict__ pmax, int N) {
  int c = threadIdx.x;
  float s = 0.f; unsigned mb = 0u;
  for (int r = blockIdx.x; r < N; r += gridDim.x) {
    float v = h[(size_t)r * D + c];
    s += v;
    unsigned e = fenc(v);
    mb = (e > mb) ? e : mb;
  }
  atomicAdd(&psum[c], s);
  atomicMax(&pmax[c], mb);
}

// pooled = [mean, max, sum] (768) -> W_pool -> relu(Wo1) -> Wo2 -> se[256]
__global__ void final_mlp_kernel(const float* __restrict__ psum, const unsigned* __restrict__ pmax,
                                 const float* __restrict__ W_pool, const float* __restrict__ b_pool,
                                 const float* __restrict__ Wo1, const float* __restrict__ bo1,
                                 const float* __restrict__ Wo2, const float* __restrict__ bo2,
                                 float* __restrict__ se, int N) {
  __shared__ float pooled[3 * D];
  __shared__ float t0[D];
  __shared__ float t1[D];
  int t = threadIdx.x;
  float s = psum[t];
  pooled[t]         = s / (float)N;
  pooled[D + t]     = fdec(pmax[t]);
  pooled[2 * D + t] = s;
  __syncthreads();
  float a = b_pool[t];
  for (int k = 0; k < 3 * D; ++k) a += pooled[k] * W_pool[(size_t)k * D + t];
  t0[t] = a;
  __syncthreads();
  float b = bo1[t];
  for (int k = 0; k < D; ++k) b += t0[k] * Wo1[(size_t)k * D + t];
  t1[t] = fmaxf(b, 0.f);
  __syncthreads();
  float c = bo2[t];
  for (int k = 0; k < D; ++k) c += t1[k] * Wo2[(size_t)k * D + t];
  se[t] = c;
}

// ---------------------------------------------------------------------------
extern "C" void kernel_launch(void* const* d_in, const int* in_sizes, int n_in,
                              void* d_out, int out_size, void* d_ws, size_t ws_size,
                              hipStream_t stream) {
  const float* x          = (const float*)d_in[0];
  const int*   edge_index = (const int*)  d_in[1];
  const float* edge_attr  = (const float*)d_in[2];
  const float* W_node     = (const float*)d_in[3];
  const float* b_node     = (const float*)d_in[4];
  const float* W_edge     = (const float*)d_in[5];
  const float* b_edge     = (const float*)d_in[6];
  const float* W1         = (const float*)d_in[7];
  const float* b1         = (const float*)d_in[8];
  const float* W2         = (const float*)d_in[9];
  const float* b2         = (const float*)d_in[10];
  const float* gamma      = (const float*)d_in[11];
  const float* beta       = (const float*)d_in[12];
  const float* W_pool     = (const float*)d_in[13];
  const float* b_pool     = (const float*)d_in[14];
  const float* Wo1        = (const float*)d_in[15];
  const float* bo1        = (const float*)d_in[16];
  const float* Wo2        = (const float*)d_in[17];
  const float* bo2        = (const float*)d_in[18];
  (void)n_in; (void)out_size; (void)ws_size;

  const int N = in_sizes[0] / DNODE;   // 50000 (multiple of 16)
  const int E = in_sizes[2] / D;       // 800000 (multiple of 16)
  const int* src = edge_index;
  const int* dst = edge_index + E;

  float* se = (float*)d_out;           // output 0: [1,256]
  float* h  = (float*)d_out + D;       // output 1: node_embeddings [N,256] built in place

  // ---- workspace layout ----
  size_t off = 0;
  auto alloc = [&](size_t bytes) -> void* {
    void* p = (char*)d_ws + off;
    off = (off + bytes + 255) & ~(size_t)255;
    return p;
  };
  float*    edge_agg = (float*)alloc((size_t)N * D * 4);       // aliased as z2
  float*    xc       = (float*)alloc((size_t)N * 2 * D * 4);   // aliased as z1
  float*    acc      = (float*)alloc((size_t)N * 2 * D * 4);
  float*    s1       = (float*)alloc(512 * 4);  float* s2 = s1 + 256;
  float*    psum     = (float*)alloc(512 * 4);  unsigned* pmax = (unsigned*)(psum + 256);
  __bf16*   Bp_node  = (__bf16*)alloc((size_t)DNODE * D * 2);
  __bf16*   Bp_edge  = (__bf16*)alloc((size_t)NLAYERS * D * D * 2);
  __bf16*   Bp_w1    = (__bf16*)alloc((size_t)NLAYERS * 2 * D * D * 2);
  __bf16*   Bp_w2    = (__bf16*)alloc((size_t)NLAYERS * D * D * 2);

  // ---- pack weights to bf16 fragment layout (tiny) ----
  pack_b_kernel<<<DNODE, 256, 0, stream>>>(W_node, Bp_node, DNODE);
  for (int l = 0; l < NLAYERS; ++l) {
    pack_b_kernel<<<D,     256, 0, stream>>>(W_edge + (size_t)l*D*D,   Bp_edge + (size_t)l*D*D,   D);
    pack_b_kernel<<<2 * D, 256, 0, stream>>>(W1     + (size_t)l*2*D*D, Bp_w1   + (size_t)l*2*D*D, 2*D);
    pack_b_kernel<<<D,     256, 0, stream>>>(W2     + (size_t)l*D*D,   Bp_w2   + (size_t)l*D*D,   D);
  }

  // ---- node encoder: h = x @ W_node + b_node ----
  gemm_wmma_n256<0, DNODE><<<N / 16, 256, 0, stream>>>(x, Bp_node, b_node, h, nullptr);

  for (int l = 0; l < NLAYERS; ++l) {
    // edge encode + fused scatter-add:  edge_agg[dst] += edge_attr@W_edge + b_edge
    hipMemsetAsync(edge_agg, 0, (size_t)N * D * 4, stream);
    gemm_wmma_n256<2, D><<<E / 16, 256, 0, stream>>>(edge_attr, Bp_edge + (size_t)l*D*D,
                                                     b_edge + l*D, edge_agg, dst);
    // xc = [h, edge_agg]; acc = xc
    build_xc_kernel<<<N, 256, 0, stream>>>(h, edge_agg, xc, acc);
    // acc[dst] += xc[src]
    gin_msg_kernel<<<E * 128 / 256, 256, 0, stream>>>(xc, src, dst, acc);
    // z1 = relu(acc @ W1 + b1); z2 = z1 @ W2 + b2   (z1 aliases xc, z2 aliases edge_agg)
    float* z1 = xc;
    float* z2 = edge_agg;
    gemm_wmma_n256<1, 2 * D><<<N / 16, 256, 0, stream>>>(acc, Bp_w1 + (size_t)l*2*D*D, b1 + l*D, z1, nullptr);
    gemm_wmma_n256<0, D><<<N / 16, 256, 0, stream>>>(z1, Bp_w2 + (size_t)l*D*D, b2 + l*D, z2, nullptr);
    // batchnorm (training stats, biased var) + residual relu
    hipMemsetAsync(s1, 0, 512 * 4, stream);
    colstat_kernel<<<512, 256, 0, stream>>>(z2, s1, s2, N);
    bn_apply_kernel<<<N, 256, 0, stream>>>(z2, s1, s2, gamma + l*D, beta + l*D, h, N);
  }

  // ---- pooling + output MLP ----
  hipMemsetAsync(psum, 0, 512 * 4, stream);
  pool_kernel<<<512, 256, 0, stream>>>(h, psum, pmax, N);
  final_mlp_kernel<<<1, 256, 0, stream>>>(psum, pmax, W_pool, b_pool, Wo1, bo1, Wo2, bo2, se, N);
}
